// SparseCrossAttention_65798898975030
// MI455X (gfx1250) — matrix operand order repaired and checked by reading, not compiled, not run on hardware
//
#include <hip/hip_runtime.h>

// ---- problem constants (from reference setup_inputs) ----
constexpr int B   = 4;
constexpr int S   = 4096;
constexpr int SKV = 1024;
constexpr int D   = 1024;
constexpr int H   = 16;
constexpr int HD  = 64;     // head dim
constexpr int TQ  = 16;     // query tile

typedef __attribute__((ext_vector_type(16))) __bf16 v16bf;
typedef __attribute__((ext_vector_type(8)))  __bf16 v8bf;
typedef __attribute__((ext_vector_type(8)))  float  v8f;
typedef __attribute__((ext_vector_type(4)))  float  v4f;

__device__ __forceinline__ v8f zero8() {
  v8f c;
#pragma unroll
  for (int i = 0; i < 8; ++i) c[i] = 0.0f;
  return c;
}

__device__ __forceinline__ v8f wmma_bf16(v16bf a, v16bf b, v8f c) {
  return __builtin_amdgcn_wmma_f32_16x16x32_bf16(false, a, false, b, (short)0, c,
                                                 false, false);
}

// A-fragment 16x32 (M=rows, K=contraction) from row-major bf16, stride ld elems.
__device__ __forceinline__ v16bf load_a_bf16(const __bf16* base, int ld, int lane) {
  int m  = lane & 15;
  int kh = (lane >> 4) << 3;
  const __bf16* p = base + (size_t)m * ld + kh;
  v8bf lo = *(const v8bf*)(p);
  v8bf hi = *(const v8bf*)(p + 16);
  v16bf a;
#pragma unroll
  for (int i = 0; i < 8; ++i) { a[i] = lo[i]; a[i + 8] = hi[i]; }
  return a;
}

// B-fragment 32x16 (K=contraction, N=cols). src holds B^T row-major:
// src[n][k] contiguous in k; lane&15 = col N, K (lane>>4)*8..+7 and 16+(lane>>4)*8..+7.
__device__ __forceinline__ v16bf load_b_bf16(const __bf16* src, int ld, int lane) {
  int n  = lane & 15;
  int kb = (lane >> 4) << 3;
  const __bf16* p = src + (size_t)n * ld + kb;
  v8bf lo = *(const v8bf*)(p);
  v8bf hi = *(const v8bf*)(p + 16);
  v16bf b;
#pragma unroll
  for (int i = 0; i < 8; ++i) { b[i] = lo[i]; b[i + 8] = hi[i]; }
  return b;
}

// ---------------- 0. one-shot fp32 -> bf16 weight conversion -----------
__global__ __launch_bounds__(256) void cvt_bf16_kernel(
    const float* __restrict__ src, __bf16* __restrict__ dst) {
  int i = (blockIdx.x * 256 + threadIdx.x) * 8;
  v4f f0 = *(const v4f*)(src + i);
  v4f f1 = *(const v4f*)(src + i + 4);
  v8bf o;
#pragma unroll
  for (int j = 0; j < 4; ++j) { o[j] = (__bf16)f0[j]; o[j + 4] = (__bf16)f1[j]; }
  *(v8bf*)(dst + i) = o;
}

// ---------------- 1. per-token scores: one wave per row ----------------
__global__ __launch_bounds__(256) void scores_kernel(
    const float* __restrict__ base, const float* __restrict__ wsc,
    const float* __restrict__ bsc, float* __restrict__ scores) {
  int row  = blockIdx.x * 8 + (threadIdx.x >> 5);
  int lane = threadIdx.x & 31;
  if (row >= B * S) return;
  const float* x = base + (size_t)row * D;
  float s = 0.0f;
  for (int i = lane; i < D; i += 32) s += x[i] * wsc[i];
#pragma unroll
  for (int m = 16; m >= 1; m >>= 1) s += __shfl_xor(s, m, 32);
  if (lane == 0) scores[row] = s + bsc[0];
}

// ---------------- 2. exact top-k by rank (unique ranks, no atomics) ----
__global__ __launch_bounds__(256) void select_kernel(
    const float* __restrict__ scores, const int* __restrict__ dk,
    int* __restrict__ sel) {
  __shared__ float sc[S];
  int b   = blockIdx.y;
  int tid = threadIdx.x;
  for (int i = tid; i < S; i += 256) sc[i] = scores[b * S + i];
  __syncthreads();
  int s = blockIdx.x * 256 + tid;
  float my = sc[s];
  int rank = 0;
  for (int t = 0; t < S; ++t) {
    float v = sc[t];
    rank += (v > my) || (v == my && t < s);
  }
  if (rank < dk[0]) sel[b * S + rank] = s;
}

// Pipelined 16-row x 64-col GEMM pass: acc += xs(16xD) @ w^T slice.
// xs: LDS bf16 [16][D]; w: bf16 [D out][D in] row-major; cols c0..c0+63.
__device__ __forceinline__ void gemm_pass(const __bf16* xs, const __bf16* w,
                                          int c0, int lane, v8f acc[4]) {
  const __bf16* wb = w + (size_t)c0 * D;
  v16bf a_c = load_a_bf16(xs, D, lane);
  v16bf b_c[4];
#pragma unroll
  for (int nb = 0; nb < 4; ++nb)
    b_c[nb] = load_b_bf16(wb + (size_t)(nb * 16) * D, D, lane);
  for (int d = 0; d < D; d += 32) {
    int dn = (d + 32 < D) ? d + 32 : 0;           // clamped prefetch addr
    v16bf a_n = load_a_bf16(xs + dn, D, lane);
    v16bf b_n[4];
#pragma unroll
    for (int nb = 0; nb < 4; ++nb)
      b_n[nb] = load_b_bf16(wb + (size_t)(nb * 16) * D + dn, D, lane);
#pragma unroll
    for (int nb = 0; nb < 4; ++nb) acc[nb] = wmma_bf16(a_c, b_c[nb], acc[nb]);
    a_c = a_n;
#pragma unroll
    for (int nb = 0; nb < 4; ++nb) b_c[nb] = b_n[nb];
  }
}

// ---------------- 3. K / V projection (bf16 WMMA) ----------------------
// Kp stored [b][skv][D] row-major; Vp stored transposed [b][D][skv].
__global__ __launch_bounds__(512) void kvproj_kernel(
    const float* __restrict__ scaf,
    const __bf16* __restrict__ wkb, const float* __restrict__ bk,
    const __bf16* __restrict__ wvb, const float* __restrict__ bv,
    __bf16* __restrict__ Kp, __bf16* __restrict__ Vpt) {
  __shared__ alignas(16) __bf16 xs[TQ * D];   // 32 KB
  int tile = blockIdx.x, b = blockIdx.y;
  int tid  = threadIdx.x;
  int wave = tid >> 5, lane = tid & 31;
  {
    const float* src = scaf + ((size_t)b * SKV + tile * TQ + wave) * D;
    for (int i = lane; i < D; i += 32) xs[wave * D + i] = (__bf16)src[i];
  }
  __syncthreads();

  int nl = lane & 15, hf = lane >> 4;
  int c0 = wave * HD;                         // 64 output cols per wave

  // pass 1: K projection
  {
    v8f ck[4];
#pragma unroll
    for (int nb = 0; nb < 4; ++nb) ck[nb] = zero8();
    gemm_pass(xs, wkb, c0, lane, ck);
#pragma unroll
    for (int nb = 0; nb < 4; ++nb) {
      float bkc = bk[c0 + nb * 16 + nl];
#pragma unroll
      for (int j = 0; j < 8; ++j) {
        int row = tile * TQ + j + 8 * hf;
        Kp[((size_t)b * SKV + row) * D + c0 + nb * 16 + nl] = (__bf16)(ck[nb][j] + bkc);
      }
    }
  }
  // pass 2: V projection (stored transposed)
  {
    v8f cv[4];
#pragma unroll
    for (int nb = 0; nb < 4; ++nb) cv[nb] = zero8();
    gemm_pass(xs, wvb, c0, lane, cv);
#pragma unroll
    for (int nb = 0; nb < 4; ++nb) {
      float bvc = bv[c0 + nb * 16 + nl];
#pragma unroll
      for (int j = 0; j < 8; ++j) {
        int row = tile * TQ + j + 8 * hf;
        Vpt[((size_t)b * D + c0 + nb * 16 + nl) * SKV + row] = (__bf16)(cv[nb][j] + bvc);
      }
    }
  }
}

// ---------------- 4. sparse flash cross-attention + out-proj -----------
__global__ __launch_bounds__(512) void attn_kernel(
    const float* __restrict__ base, const int* __restrict__ sel,
    const int* __restrict__ dk,
    const __bf16* __restrict__ Kp, const __bf16* __restrict__ Vpt,
    const __bf16* __restrict__ wqb, const float* __restrict__ bq,
    const __bf16* __restrict__ wob, const float* __restrict__ bo,
    float* __restrict__ out) {
  __shared__ alignas(16) __bf16 xq[TQ * D];        // 32 KB; reused as ctx
  __shared__ alignas(16) __bf16 stg[16 * TQ * 32]; // 16 KB; 16x32 per wave
  int tile = blockIdx.x, b = blockIdx.y;
  int kval = dk[0];
  if (tile * TQ >= kval) return;
  int tid  = threadIdx.x;
  int wave = tid >> 5, lane = tid & 31;
  int nl = lane & 15, hf = lane >> 4;
  int h  = wave;                                   // one wave per head

  // gather top-k rows (clamped -> duplicates for partial tile, not stored)
  {
    int grow = sel[b * S + min(tile * TQ + wave, kval - 1)];
    const float* src = base + ((size_t)b * S + grow) * D;
    for (int i = lane; i < D; i += 32) xq[wave * D + i] = (__bf16)src[i];
  }
  __syncthreads();

  // ---- Q projection: q = x @ wq^T + bq, scaled by 1/sqrt(hd) ----
  v8f cq[4];
#pragma unroll
  for (int nb = 0; nb < 4; ++nb) cq[nb] = zero8();
  gemm_pass(xq, wqb, h * HD, lane, cq);

  // C-layout -> A-layout transpose through per-wave LDS staging (two passes)
  __bf16* qst = stg + wave * TQ * 32;
  const float qscale = 0.125f;                     // 1/sqrt(64)
  v16bf Aq[2];
#pragma unroll
  for (int pass = 0; pass < 2; ++pass) {
#pragma unroll
    for (int n2 = 0; n2 < 2; ++n2) {
      int nb = pass * 2 + n2;
      float bias = bq[h * HD + nb * 16 + nl];
#pragma unroll
      for (int j = 0; j < 8; ++j)
        qst[(j + 8 * hf) * 32 + n2 * 16 + nl] = (__bf16)((cq[nb][j] + bias) * qscale);
    }
    Aq[pass] = load_a_bf16(qst, 32, lane);         // K = hd slice pass*32..+31
  }
  __syncthreads();                                 // xq free -> reuse as ctx
  __bf16* ctxs = xq;

  // ---- flash attention over 32-key chunks, software-pipelined ----
  float m[8], l[8];
  v8f cctx[4];
#pragma unroll
  for (int j = 0; j < 8; ++j) { m[j] = -INFINITY; l[j] = 0.0f; }
#pragma unroll
  for (int nb = 0; nb < 4; ++nb) cctx[nb] = zero8();

  const __bf16* KpB = Kp + (size_t)b * SKV * D + h * HD;
  const __bf16* VpB = Vpt + ((size_t)b * D + h * HD) * SKV;
  __bf16* pst = qst;                               // reuse staging tile

  // preload K fragments for k0 = 0 (kf[2*nb+c] = keys nb*16, hd c*32)
  v16bf kf[4];
#pragma unroll
  for (int nb = 0; nb < 2; ++nb) {
    kf[2 * nb]     = load_b_bf16(KpB + (size_t)(nb * 16) * D, D, lane);
    kf[2 * nb + 1] = load_b_bf16(KpB + (size_t)(nb * 16) * D + 32, D, lane);
  }

  for (int k0 = 0; k0 < SKV; k0 += 32) {
    // QK^T: 16q x 32keys (two 16x16 C tiles), contraction hd=64
    v8f cl[2];
    cl[0] = zero8(); cl[1] = zero8();
#pragma unroll
    for (int nb = 0; nb < 2; ++nb) {
      cl[nb] = wmma_bf16(Aq[0], kf[2 * nb], cl[nb]);
      cl[nb] = wmma_bf16(Aq[1], kf[2 * nb + 1], cl[nb]);
    }
    // prefetch next iteration's K fragments (clamped address, uniform)
    int k0n = (k0 + 32 < SKV) ? k0 + 32 : 0;
#pragma unroll
    for (int nb = 0; nb < 2; ++nb) {
      kf[2 * nb]     = load_b_bf16(KpB + (size_t)(k0n + nb * 16) * D, D, lane);
      kf[2 * nb + 1] = load_b_bf16(KpB + (size_t)(k0n + nb * 16) * D + 32, D, lane);
    }
    // online softmax (latency of K/V loads hidden behind this VALU work)
    float pr[2][8];
#pragma unroll
    for (int j = 0; j < 8; ++j) {
      float vmax = fmaxf(cl[0][j], cl[1][j]);
#pragma unroll
      for (int x = 1; x < 16; x <<= 1) vmax = fmaxf(vmax, __shfl_xor(vmax, x, 32));
      float mn   = fmaxf(m[j], vmax);
      float corr = __expf(m[j] - mn);
      float p0   = __expf(cl[0][j] - mn);
      float p1   = __expf(cl[1][j] - mn);
      float rs   = p0 + p1;
#pragma unroll
      for (int x = 1; x < 16; x <<= 1) rs += __shfl_xor(rs, x, 32);
      l[j] = l[j] * corr + rs;
      m[j] = mn;
      pr[0][j] = p0; pr[1][j] = p1;
#pragma unroll
      for (int nb = 0; nb < 4; ++nb) cctx[nb][j] *= corr;
    }
    // issue V loads early; their latency is hidden by the LDS transpose
    v16bf vf[4];
#pragma unroll
    for (int nb = 0; nb < 4; ++nb)
      vf[nb] = load_b_bf16(VpB + (size_t)(nb * 16) * SKV + k0, SKV, lane);
    // P (C-layout) -> A-layout via LDS
#pragma unroll
    for (int nb = 0; nb < 2; ++nb)
#pragma unroll
      for (int j = 0; j < 8; ++j)
        pst[(j + 8 * hf) * 32 + nb * 16 + nl] = (__bf16)pr[nb][j];
    v16bf Ap = load_a_bf16(pst, 32, lane);
#pragma unroll
    for (int nb = 0; nb < 4; ++nb) cctx[nb] = wmma_bf16(Ap, vf[nb], cctx[nb]);
  }

  // normalize and drop head ctx into shared concat buffer (bf16)
#pragma unroll
  for (int j = 0; j < 8; ++j) l[j] = 1.0f / l[j];
#pragma unroll
  for (int nb = 0; nb < 4; ++nb)
#pragma unroll
    for (int j = 0; j < 8; ++j)
      ctxs[(j + 8 * hf) * D + h * HD + nb * 16 + nl] = (__bf16)(cctx[nb][j] * l[j]);
  __syncthreads();

  // ---- output projection: out = ctx @ wo^T + bo, scattered to top-k rows
  v8f co[4];
#pragma unroll
  for (int nb = 0; nb < 4; ++nb) co[nb] = zero8();
  int c0 = wave * HD;
  gemm_pass(ctxs, wob, c0, lane, co);
#pragma unroll
  for (int nb = 0; nb < 4; ++nb) {
    float bias = bo[c0 + nb * 16 + nl];
#pragma unroll
    for (int j = 0; j < 8; ++j) {
      int qi = tile * TQ + j + 8 * hf;
      if (qi < kval) {
        int grow = sel[b * S + qi];
        out[((size_t)b * S + grow) * D + c0 + nb * 16 + nl] = co[nb][j] + bias;
      }
    }
  }
}

// ---------------- host-side launch ----------------
extern "C" void kernel_launch(void* const* d_in, const int* in_sizes, int n_in,
                              void* d_out, int out_size, void* d_ws, size_t ws_size,
                              hipStream_t stream) {
  const float* base = (const float*)d_in[0];
  const float* scaf = (const float*)d_in[1];
  const float* wsc  = (const float*)d_in[2];
  const float* bsc  = (const float*)d_in[3];
  const float* wq   = (const float*)d_in[4];
  const float* wk   = (const float*)d_in[5];
  const float* wv   = (const float*)d_in[6];
  const float* bq   = (const float*)d_in[7];
  const float* bk   = (const float*)d_in[8];
  const float* bv   = (const float*)d_in[9];
  const float* wo   = (const float*)d_in[10];
  const float* bo   = (const float*)d_in[11];
  // d_in[12] = sparsity_factor (folded into k already)
  const int* dk     = (const int*)d_in[13];

  char* ws = (char*)d_ws;
  float*  scores = (float*)ws;                              // 64 KB
  int*    sel    = (int*)(ws + (size_t)64 * 1024);          // 64 KB
  __bf16* Kp     = (__bf16*)(ws + (size_t)128 * 1024);      // 8 MB
  __bf16* Vpt    = Kp  + (size_t)B * SKV * D;               // 8 MB
  __bf16* wqb    = Vpt + (size_t)B * SKV * D;               // 2 MB each
  __bf16* wkb    = wqb + (size_t)D * D;
  __bf16* wvb    = wkb + (size_t)D * D;
  __bf16* wob    = wvb + (size_t)D * D;

  float* out = (float*)d_out;

  // 0. baseline copy: out = base_hidden (scatter target)
  hipMemcpyAsync(out, base, (size_t)B * S * D * sizeof(float),
                 hipMemcpyDeviceToDevice, stream);
  // 1. weights -> bf16 (once per launch; tiny vs GEMM cost)
  int cvtBlocks = (D * D) / (256 * 8);
  cvt_bf16_kernel<<<cvtBlocks, 256, 0, stream>>>(wq, wqb);
  cvt_bf16_kernel<<<cvtBlocks, 256, 0, stream>>>(wk, wkb);
  cvt_bf16_kernel<<<cvtBlocks, 256, 0, stream>>>(wv, wvb);
  cvt_bf16_kernel<<<cvtBlocks, 256, 0, stream>>>(wo, wob);
  // 2. scores
  scores_kernel<<<(B * S) / 8, 256, 0, stream>>>(base, wsc, bsc, scores);
  // 3. top-k selection by rank
  select_kernel<<<dim3(S / 256, B), 256, 0, stream>>>(scores, dk, sel);
  // 4. K/V projection
  kvproj_kernel<<<dim3(SKV / TQ, B), 512, 0, stream>>>(scaf, wkb, bk, wvb, bv, Kp, Vpt);
  // 5. attention + output projection, scattered into out (max tiles; early exit)
  attn_kernel<<<dim3(S / TQ, B), 512, 0, stream>>>(base, sel, dk, Kp, Vpt,
                                                   wqb, bq, wob, bo, out);
}
